// GAT_18992345383446
// MI455X (gfx1250) — matrix operand order repaired and checked by reading, not compiled, not run on hardware
//
#include <hip/hip_runtime.h>

// ---------------- problem constants (match reference) ----------------
constexpr int kN     = 4096;   // nodes
constexpr int kInF   = 512;    // input features
constexpr int kOutF  = 1024;   // output features (HEADS * HEAD_F)
constexpr int kHeads = 8;
constexpr int kHF    = 128;    // head features
#define NEG_SLOPE 0.2f

typedef __attribute__((ext_vector_type(16))) _Float16 v16h;
typedef __attribute__((ext_vector_type(8)))  _Float16 v8h;
typedef __attribute__((ext_vector_type(4)))  _Float16 v4h;
typedef __attribute__((ext_vector_type(8)))  float    v8f;
typedef __attribute__((ext_vector_type(4)))  float    v4f;

__device__ __forceinline__ v16h cat8(v8h a, v8h b) {
  return __builtin_shufflevector(a, b, 0,1,2,3,4,5,6,7,8,9,10,11,12,13,14,15);
}

// =====================================================================
// Kernel 0: precision/layout prep.
//   xf  [kN][kInF]    f16 row-major            = x
//   wt  [kOutF][kInF] f16 (o-major, K contig)  : wt[o][m] = W[h][m][f], o=h*128+f
//   wrt [kOutF][kInF] f16 (o-major, K contig)  : wrt[o][m] = W_r[m][o]
// =====================================================================
__global__ void gat_prep_kernel(const float* __restrict__ x,
                                const float* __restrict__ W,
                                const float* __restrict__ Wr,
                                _Float16* __restrict__ xf,
                                _Float16* __restrict__ wt,
                                _Float16* __restrict__ wrt) {
  int i = blockIdx.x * blockDim.x + threadIdx.x;
  if (i < kN * kInF) xf[i] = (_Float16)x[i];
  if (i < kOutF * kInF) {
    int o = i / kInF, m = i % kInF;
    int h = o / kHF,  f = o % kHF;
    wt[i]  = (_Float16)W[(h * kInF + m) * kHF + f];
    wrt[i] = (_Float16)Wr[m * kOutF + o];
  }
}

// =====================================================================
// Kernels 1/2: WMMA GEMM  C[n][o] = sum_k A[n][k] * B[o][k]
//   A: xf [kN][kInF] f16.  B: [kOutF][kInF] f16 (K contiguous per row).
// Each wave computes a 16x32 register-blocked tile (one A fragment feeds
// two v_wmma_f32_16x16x32_f16). 8 waves/block -> block covers 16x256.
//   MODE 0: store C transposed as f16 -> ht[o][kN]   (attention B operand)
//   MODE 1: store C as f32 row-major + bias -> out   (residual path)
// =====================================================================
template <int MODE>
__global__ void gat_gemm_kernel(const _Float16* __restrict__ A,
                                const _Float16* __restrict__ B,
                                _Float16* __restrict__ Ct,
                                float* __restrict__ Cp,
                                const float* __restrict__ bias) {
  const int lane = threadIdx.x & 31;
  const int wave = threadIdx.x >> 5;                 // 0..7
  const int lrow = lane & 15;                        // M or N within tile
  const int lhi  = lane >> 4;                        // K-half selector
  const int n0   = blockIdx.x * 16;
  const int o0   = (blockIdx.y * 8 + wave) * 32;     // 32 output cols per wave

  // A fragment (16x32 f16): lane holds row n0+lrow, K = k0 + lhi*16 + {0..15}
  const _Float16* ap  = A + (size_t)(n0 + lrow) * kInF + lhi * 16;
  // B fragments (32x16 f16): lane holds col, K = k0 + lhi*16 + {0..15}
  const _Float16* bp0 = B + (size_t)(o0 + lrow) * kInF + lhi * 16;
  const _Float16* bp1 = B + (size_t)(o0 + 16 + lrow) * kInF + lhi * 16;

  v8f acc0 = {}, acc1 = {};
#pragma unroll 4
  for (int k0 = 0; k0 < kInF; k0 += 32) {
    v16h a  = *(const v16h*)(ap + k0);
    v16h b0 = *(const v16h*)(bp0 + k0);
    v16h b1 = *(const v16h*)(bp1 + k0);
    acc0 = __builtin_amdgcn_wmma_f32_16x16x32_f16(false, a, false, b0,
                                                  (short)0, acc0, false, false);
    acc1 = __builtin_amdgcn_wmma_f32_16x16x32_f16(false, a, false, b1,
                                                  (short)0, acc1, false, false);
  }

  if (MODE == 0) {
    // C/D layout: row = j + 8*lhi, col = lrow -> transposed store contiguous
    _Float16* cp0 = Ct + (size_t)(o0 + lrow) * kN + n0 + 8 * lhi;
    _Float16* cp1 = Ct + (size_t)(o0 + 16 + lrow) * kN + n0 + 8 * lhi;
    v8h h0, h1;
#pragma unroll
    for (int j = 0; j < 8; ++j) { h0[j] = (_Float16)acc0[j]; h1[j] = (_Float16)acc1[j]; }
    *(v8h*)cp0 = h0;
    *(v8h*)cp1 = h1;
  } else {
    float bv0 = bias[o0 + lrow];
    float bv1 = bias[o0 + 16 + lrow];
#pragma unroll
    for (int j = 0; j < 8; ++j) {
      const size_t row = (size_t)(n0 + j + 8 * lhi) * kOutF;
      Cp[row + o0 + lrow]      = acc0[j] + bv0;
      Cp[row + o0 + 16 + lrow] = acc1[j] + bv1;
    }
  }
}

// =====================================================================
// Kernel 3: per-(head,node) attention coefficients
//   ci[h][n] = h[h,n,:] . w_i[h],   cj[h][n] = h[h,n,:] . w_j[h]
// Reads ht[o][n] -> coalesced across consecutive n.
// =====================================================================
__global__ void gat_coe_kernel(const _Float16* __restrict__ ht,
                               const float* __restrict__ wi,
                               const float* __restrict__ wj,
                               float* __restrict__ ci, float* __restrict__ cj) {
  int idx = blockIdx.x * blockDim.x + threadIdx.x;   // h*kN + n
  if (idx >= kHeads * kN) return;
  int h = idx / kN, n = idx % kN;
  float si = 0.f, sj = 0.f;
#pragma unroll 8
  for (int f = 0; f < kHF; ++f) {
    float hv = (float)ht[(size_t)(h * kHF + f) * kN + n];
    si += hv * wi[h * kHF + f];
    sj += hv * wj[h * kHF + f];
  }
  ci[idx] = si;
  cj[idx] = sj;
}

// =====================================================================
// Kernel 4: softmax row statistics (max + 1/denominator) per (head, node).
// Block = 256 threads handles a 16-node tile: thread (tn,tm) strides m by 16.
// mask[n][m] = graph.T[n][m] = graph[m][n]  (coalesced: 16 consecutive n).
// =====================================================================
__global__ void gat_stats_kernel(const int* __restrict__ graph,
                                 const float* __restrict__ ci,
                                 const float* __restrict__ cj,
                                 float* __restrict__ rowmax,
                                 float* __restrict__ rdenom) {
  const int h  = blockIdx.y;
  const int n0 = blockIdx.x * 16;
  const int tn = threadIdx.x & 15;
  const int tm = threadIdx.x >> 4;                   // 0..15
  const int n  = n0 + tn;
  const float civ = ci[h * kN + n];
  const float* cjh = cj + h * kN;

  __shared__ float red[16][17];
  __shared__ float rm[16];

  float lmax = -INFINITY;
  for (int m = tm; m < kN; m += 16) {
    float v = civ + cjh[m];
    v = fmaxf(v, NEG_SLOPE * v);                     // leaky_relu (slope<1)
    lmax = graph[(size_t)m * kN + n] > 0 ? fmaxf(lmax, v) : lmax;
  }
  red[tm][tn] = lmax;
  __syncthreads();
  if (tm == 0) {
    float mx = red[0][tn];
#pragma unroll
    for (int k = 1; k < 16; ++k) mx = fmaxf(mx, red[k][tn]);
    rm[tn] = mx;
    rowmax[h * kN + n] = mx;
  }
  __syncthreads();
  const float mx = rm[tn];
  float lsum = 0.f;
  for (int m = tm; m < kN; m += 16) {
    float v = civ + cjh[m];
    v = fmaxf(v, NEG_SLOPE * v);
    float e = __expf(v - mx);
    lsum += graph[(size_t)m * kN + n] > 0 ? e : 0.0f;
  }
  __syncthreads();
  red[tm][tn] = lsum;
  __syncthreads();
  if (tm == 0) {
    float s = 0.f;
#pragma unroll
    for (int k = 0; k < 16; ++k) s += red[k][tn];
    rdenom[h * kN + n] = s > 0.f ? 1.0f / s : 0.0f;
  }
}

// =====================================================================
// Kernel 5: streaming attention * h  (the 34-GFLOP GEMM), flash-style.
// Block = 256 thr = 8 waves, per (head, 16-row n-tile). Output tile 16x128.
// K-chunk = 64: all threads branchlessly build the 16x64 f16 probability
// tile (4 adjacent slots per thread, one packed ds_store_b64), then each
// wave issues two v_wmma against its 16-column slice of ht (K contiguous).
// Accumulate f32, add into d_out (which already holds residual + bias).
// =====================================================================
__global__ void gat_attn_kernel(const int* __restrict__ graph,
                                const float* __restrict__ ci,
                                const float* __restrict__ cj,
                                const float* __restrict__ rowmax,
                                const float* __restrict__ rdenom,
                                const _Float16* __restrict__ ht,
                                float* __restrict__ out) {
  const int h  = blockIdx.y;
  const int n0 = blockIdx.x * 16;
  const int t    = threadIdx.x;
  const int lane = t & 31;
  const int wave = t >> 5;                           // 0..7 -> 16-col f slice
  const int lrow = lane & 15;
  const int lhi  = lane >> 4;

  __shared__ __align__(32) _Float16 pl[16][64];      // p[n-in-tile][m-in-chunk]
  __shared__ float s_ci[16], s_rm[16], s_rd[16];

  if (t < 16) {
    s_ci[t] = ci[h * kN + n0 + t];
    s_rm[t] = rowmax[h * kN + n0 + t];
    s_rd[t] = rdenom[h * kN + n0 + t];
  }
  __syncthreads();

  const int nn   = t & 15;                           // node within tile
  const int quad = t >> 4;                           // 4-slot group (0..15)
  const float civ = s_ci[nn];
  const float rmv = s_rm[nn];
  const float rdv = s_rd[nn];
  const float* cjh = cj + h * kN;
  const int* gp = graph + n0 + nn;                   // + m*kN per neighbor

  // B operand: ht rows are K(m)-contiguous. lane -> col f = wave*16+lrow,
  // K = m0 + lhi*16 + {0..15}
  const _Float16* bp = ht + (size_t)(h * kHF + wave * 16 + lrow) * kN + lhi * 16;

  v8f acc = {};
  for (int m0 = 0; m0 < kN; m0 += 64) {
    // ---- branchless probability quad (adjacent slots, packed store) ----
    const int mq = m0 + quad * 4;
    v4f cjv = *(const v4f*)(cjh + mq);
    v4h pv;
#pragma unroll
    for (int jj = 0; jj < 4; ++jj) {
      float v = civ + cjv[jj];
      v = fmaxf(v, NEG_SLOPE * v);                   // leaky_relu
      float e = __expf(v - rmv) * rdv;               // softmax weight
      pv[jj] = (_Float16)(gp[(size_t)(mq + jj) * kN] > 0 ? e : 0.0f);
    }
    *(v4h*)(&pl[nn][quad * 4]) = pv;
    __builtin_prefetch(bp + m0 + 64, 0, 1);          // next hT chunk (L2)
    __syncthreads();

    // ---- two WMMAs per barrier pair: A from LDS, B from hT (L2) ----
    const _Float16* prow = &pl[lrow][lhi * 16];
    v16h a0 = *(const v16h*)(prow);
    v16h a1 = *(const v16h*)(prow + 32);
    v16h b0 = *(const v16h*)(bp + m0);
    v16h b1 = *(const v16h*)(bp + m0 + 32);
    acc = __builtin_amdgcn_wmma_f32_16x16x32_f16(false, a0, false, b0,
                                                 (short)0, acc, false, false);
    acc = __builtin_amdgcn_wmma_f32_16x16x32_f16(false, a1, false, b1,
                                                 (short)0, acc, false, false);
    __syncthreads();   // LDS tile consumed; safe to overwrite next chunk
  }

  const int o = h * kHF + wave * 16 + lrow;
#pragma unroll
  for (int j = 0; j < 8; ++j) {
    const int n = n0 + j + 8 * lhi;
    out[(size_t)n * kOutF + o] += acc[j];            // residual+bias already there
  }
}

// =====================================================================
// Launcher
// =====================================================================
extern "C" void kernel_launch(void* const* d_in, const int* in_sizes, int n_in,
                              void* d_out, int out_size, void* d_ws, size_t ws_size,
                              hipStream_t stream) {
  const float* x     = (const float*)d_in[0];   // [4096,512]
  const int*   graph = (const int*)  d_in[1];   // [4096,4096]
  const float* W     = (const float*)d_in[2];   // [8,512,128]
  const float* wi    = (const float*)d_in[3];   // [8,128,1]
  const float* wj    = (const float*)d_in[4];   // [8,128,1]
  const float* Wr    = (const float*)d_in[5];   // [512,1024]
  const float* bias  = (const float*)d_in[6];   // [1024]
  float* out = (float*)d_out;                   // [4096,1024]

  // ---- workspace carve-up (~14.5 MB) ----
  char* ws = (char*)d_ws;
  size_t off = 0;
  auto take = [&](size_t bytes) { char* p = ws + off; off += (bytes + 255) & ~size_t(255); return p; };
  _Float16* xf  = (_Float16*)take((size_t)kN * kInF * 2);
  _Float16* wt  = (_Float16*)take((size_t)kOutF * kInF * 2);
  _Float16* wrt = (_Float16*)take((size_t)kOutF * kInF * 2);
  _Float16* ht  = (_Float16*)take((size_t)kOutF * kN * 2);
  float* ci = (float*)take((size_t)kHeads * kN * 4);
  float* cj = (float*)take((size_t)kHeads * kN * 4);
  float* rm = (float*)take((size_t)kHeads * kN * 4);
  float* rd = (float*)take((size_t)kHeads * kN * 4);
  (void)ws_size; (void)n_in; (void)in_sizes; (void)out_size;

  // 0) f16 conversion + weight transposes
  {
    int total = kN * kInF;                       // >= kOutF*kInF
    gat_prep_kernel<<<dim3((total + 255) / 256), dim3(256), 0, stream>>>(
        x, W, Wr, xf, wt, wrt);
  }
  // 1) hT = (x @ W)^T  as f16 [1024][4096]
  gat_gemm_kernel<0><<<dim3(kN / 16, kOutF / 256), dim3(256), 0, stream>>>(
      xf, wt, ht, nullptr, bias);
  // 2) out = x @ W_r + bias  (residual base, f32)
  gat_gemm_kernel<1><<<dim3(kN / 16, kOutF / 256), dim3(256), 0, stream>>>(
      xf, wrt, nullptr, out, bias);
  // 3) attention coefficients
  gat_coe_kernel<<<dim3(kHeads * kN / 256), dim3(256), 0, stream>>>(
      ht, wi, wj, ci, cj);
  // 4) softmax row stats (max, 1/denom)
  gat_stats_kernel<<<dim3(kN / 16, kHeads), dim3(256), 0, stream>>>(
      graph, ci, cj, rm, rd);
  // 5) out += softmax(coe+mask) @ h
  gat_attn_kernel<<<dim3(kN / 16, kHeads), dim3(256), 0, stream>>>(
      graph, ci, cj, rm, rd, ht, out);
}
